// OffsetBlock_78761110274182
// MI455X (gfx1250) — matrix operand order repaired and checked by reading, not compiled, not run on hardware
//
#include <hip/hip_runtime.h>
#include <math.h>

// ---------------- problem constants (match reference) ----------------
#define Bsz 16
#define CIN 256
#define Hh 64
#define Ww 64
#define HWp 4096            // 64*64
#define DP 512
#define COUT 256
#define EPSF 1.1920929e-07f
#define BN_EPS 1e-5f
#define EXP_CLAMP 88.722835f

typedef __attribute__((ext_vector_type(16))) __bf16 v16bf;
typedef __attribute__((ext_vector_type(8)))  float v8f;

__device__ __forceinline__ unsigned short f2bf(float f) {
  unsigned u = __float_as_uint(f);
  u += 0x7FFFu + ((u >> 16) & 1u);          // round-to-nearest-even
  return (unsigned short)(u >> 16);
}

// ---------------------------------------------------------------------
// GEMM: Out[b,n,hw] = sum_k X[b,k,hw] * W[n,k] + bias[n] (+ skip[b,n,hw])
// M = B*HW (spatial), tiled 128; N tiled 64; K stepped 32 in bf16 WMMA.
// ---------------------------------------------------------------------
#define BLK_M 128
#define BLK_N 64
#define BLK_K 32

__global__ __launch_bounds__(256) void gemm_bf16_kernel(
    const float* __restrict__ X, const float* __restrict__ Wt,
    const float* __restrict__ bias, const float* __restrict__ skip,
    float* __restrict__ Out, int K, int Ntot)
{
  __shared__ unsigned short lds_a[BLK_M][BLK_K + 8];   // [m][k], padded
  __shared__ unsigned short lds_w[BLK_N][BLK_K + 8];   // [n][k], padded

  const int tid  = threadIdx.x;
  const int lane = tid & 31;
  const int wid  = tid >> 5;
  const int waveM = wid & 3;            // 4 waves along M
  const int waveN = wid >> 2;           // 2 waves along N
  const int m0  = blockIdx.x * BLK_M;   // global spatial-row base
  const int n0  = blockIdx.y * BLK_N;
  const int b   = m0 / HWp;             // BLK_M divides HW -> tile stays in one batch
  const int hw0 = m0 % HWp;
  const int rowInLane = lane >> 4;      // 0/1
  const int colInLane = lane & 15;

  v8f acc[2][2] = {};

  const int kIters = K / BLK_K;
  for (int kt = 0; kt < kIters; ++kt) {
    const int k0 = kt * BLK_K;
    if (kt + 1 < kIters) {
      __builtin_prefetch(X + ((size_t)b * K + (k0 + BLK_K)) * HWp + hw0, 0, 1);
      __builtin_prefetch(Wt + (size_t)n0 * K + (k0 + BLK_K), 0, 1);
    }
    // A tile: 32 k-rows x 128 m, 1024 float4 total -> 4 per thread
    #pragma unroll
    for (int i = 0; i < 4; ++i) {
      const int idx  = tid + i * 256;         // 0..1023
      const int kRow = idx >> 5;              // 0..31
      const int mq   = (idx & 31) << 2;       // 0..124
      const float4 v = *reinterpret_cast<const float4*>(
          X + ((size_t)b * K + (size_t)(k0 + kRow)) * HWp + hw0 + mq);
      lds_a[mq + 0][kRow] = f2bf(v.x);
      lds_a[mq + 1][kRow] = f2bf(v.y);
      lds_a[mq + 2][kRow] = f2bf(v.z);
      lds_a[mq + 3][kRow] = f2bf(v.w);
    }
    // W tile: 64 n-rows x 32 k, 512 float4 -> 2 per thread
    #pragma unroll
    for (int i = 0; i < 2; ++i) {
      const int idx = tid + i * 256;          // 0..511
      const int n   = idx >> 3;               // 0..63
      const int kq  = (idx & 7) << 2;         // 0..28
      const float4 v = *reinterpret_cast<const float4*>(
          Wt + (size_t)(n0 + n) * K + (k0 + kq));
      lds_w[n][kq + 0] = f2bf(v.x);
      lds_w[n][kq + 1] = f2bf(v.y);
      lds_w[n][kq + 2] = f2bf(v.z);
      lds_w[n][kq + 3] = f2bf(v.w);
    }
    __syncthreads();

    #pragma unroll
    for (int mt = 0; mt < 2; ++mt) {
      // A fragment (16x32 bf16): lane L holds row M=(L&15);
      // lanes 0-15: K {0..7,16..23}, lanes 16-31: K {8..15,24..31}, pairs/VGPR.
      union { v16bf v; unsigned u[8]; } afrag;
      const int ar = waveM * 32 + mt * 16 + colInLane;
      #pragma unroll
      for (int j = 0; j < 4; ++j)
        afrag.u[j] = *reinterpret_cast<const unsigned*>(
            &lds_a[ar][rowInLane * 8 + 2 * j]);
      #pragma unroll
      for (int j = 0; j < 4; ++j)
        afrag.u[4 + j] = *reinterpret_cast<const unsigned*>(
            &lds_a[ar][16 + rowInLane * 8 + 2 * j]);

      #pragma unroll
      for (int nt = 0; nt < 2; ++nt) {
        // B fragment (32x16 bf16): col N=(L&15); lanes 0-15 K {0..7,16..23},
        // lanes 16-31 K {8..15,24..31}, consecutive K pairs per VGPR.
        union { v16bf v; unsigned u[8]; } bfrag;
        const int bc = waveN * 32 + nt * 16 + colInLane;
        #pragma unroll
        for (int j = 0; j < 8; ++j)
          bfrag.u[j] = *reinterpret_cast<const unsigned*>(
              &lds_w[bc][rowInLane * 8 + (j >> 2) * 16 + 2 * (j & 3)]);

        acc[mt][nt] = __builtin_amdgcn_wmma_f32_16x16x32_bf16(
            false, afrag.v, false, bfrag.v, (short)0, acc[mt][nt], false, false);
      }
    }
    __syncthreads();
  }

  // Epilogue: C/D layout -> VGPR r: M = r + 8*(lane>>4), N = lane&15.
  #pragma unroll
  for (int mt = 0; mt < 2; ++mt) {
    #pragma unroll
    for (int nt = 0; nt < 2; ++nt) {
      const int nGlob = n0 + waveN * 32 + nt * 16 + colInLane;
      const float bv = bias[nGlob];
      #pragma unroll
      for (int r = 0; r < 8; ++r) {
        const int mLoc = waveM * 32 + mt * 16 + rowInLane * 8 + r;
        const size_t oi = ((size_t)b * Ntot + nGlob) * HWp + hw0 + mLoc;
        float v = acc[mt][nt][r] + bv;
        if (skip) v += skip[oi];
        Out[oi] = v;
      }
    }
  }
}

// ---------------------------------------------------------------------
// Dynamic pool: per-channel 5x5 Gaussian depthwise conv in exp space.
// One block per (b,c) plane; plane staged in LDS.
// ---------------------------------------------------------------------
__global__ __launch_bounds__(256) void dyn_pool_kernel(
    const float* __restrict__ in, const float* __restrict__ sigma,
    float* __restrict__ out)
{
  __shared__ float splane[HWp];
  __shared__ float skern[25];
  const int p = blockIdx.x;              // b*DP + c
  const int c = p % DP;
  const int tid = threadIdx.x;
  const float* ip = in + (size_t)p * HWp;
  #pragma unroll
  for (int i = 0; i < 16; ++i) {
    const int idx = tid + i * 256;
    splane[idx] = __expf(fminf(ip[idx], EXP_CLAMP));
  }
  if (tid < 25) {
    float s2 = sigma[c]; s2 = s2 * s2 + EPSF;
    const float ry = (float)(tid / 5 - 2), rx = (float)(tid % 5 - 2);
    skern[tid] = __expf(-(ry * ry + rx * rx) * 0.5f / s2);
  }
  __syncthreads();
  float ks = 0.f;
  #pragma unroll
  for (int j = 0; j < 25; ++j) ks += skern[j];
  const float kinv = 1.f / ks;

  float* op = out + (size_t)p * HWp;
  #pragma unroll
  for (int i = 0; i < 16; ++i) {
    const int idx = tid + i * 256;
    const int y = idx >> 6, x = idx & 63;
    float acc = 0.f;
    #pragma unroll
    for (int dy = -2; dy <= 2; ++dy) {
      const int yy = y + dy;
      if (yy < 0 || yy >= Hh) continue;
      #pragma unroll
      for (int dx = -2; dx <= 2; ++dx) {
        const int xx = x + dx;
        if (xx < 0 || xx >= Ww) continue;
        acc += skern[(dy + 2) * 5 + (dx + 2)] * splane[yy * Ww + xx];
      }
    }
    op[idx] = __logf(acc * kinv + EPSF);
  }
}

// ---------------------------------------------------------------------
// Displace: per-channel uniform bilinear shift with zero padding.
// ---------------------------------------------------------------------
__global__ void displace_kernel(const float* __restrict__ in,
                                const float* __restrict__ off,
                                float* __restrict__ out)
{
  const long total = (long)Bsz * DP * HWp;
  for (long i = (long)blockIdx.x * blockDim.x + threadIdx.x; i < total;
       i += (long)gridDim.x * blockDim.x) {
    const int x = (int)(i & 63);
    const int y = (int)((i >> 6) & 63);
    const long plane = i >> 12;          // b*DP + c
    const int c = (int)(plane & (DP - 1));
    const int o = c >> 3;                // 8 channels per offset
    const float dy = off[o * 2 + 0], dx = off[o * 2 + 1];
    const float y0 = floorf(dy), x0 = floorf(dx);
    const float fy = dy - y0, fx = dx - x0;
    const int yi = y + (int)y0, xi = x + (int)x0;
    const float* pp = in + plane * HWp;
    auto tap = [&](int ry, int rx) -> float {
      return (ry >= 0 && ry < Hh && rx >= 0 && rx < Ww) ? pp[ry * Ww + rx] : 0.f;
    };
    out[i] = (1.f - fy) * (1.f - fx) * tap(yi, xi)
           + (1.f - fy) * fx         * tap(yi, xi + 1)
           + fy         * (1.f - fx) * tap(yi + 1, xi)
           + fy         * fx         * tap(yi + 1, xi + 1);
  }
}

// ---------------------------------------------------------------------
// Per-channel mean/var over (B, HW). One block per channel.
// ---------------------------------------------------------------------
__global__ __launch_bounds__(256) void chan_stats_kernel(
    const float* __restrict__ in, int C,
    float* __restrict__ mean, float* __restrict__ var)
{
  __shared__ float s1[256], s2[256];
  const int c = blockIdx.x;
  const int tid = threadIdx.x;
  float s = 0.f, q = 0.f;
  for (int b = 0; b < Bsz; ++b) {
    const float* p = in + ((size_t)b * C + c) * HWp;
    for (int i = tid; i < HWp; i += 256) { const float v = p[i]; s += v; q += v * v; }
  }
  s1[tid] = s; s2[tid] = q; __syncthreads();
  for (int st = 128; st > 0; st >>= 1) {
    if (tid < st) { s1[tid] += s1[tid + st]; s2[tid] += s2[tid + st]; }
    __syncthreads();
  }
  if (tid == 0) {
    const float inv = 1.f / (float)(Bsz * HWp);
    const float m = s1[0] * inv;
    mean[c] = m;
    var[c]  = s2[0] * inv - m * m;
  }
}

// ---------------------------------------------------------------------
// BN + softplus + per-(b,c) spatial sum. One block per (b,c) plane.
// ---------------------------------------------------------------------
__global__ __launch_bounds__(256) void atten_norm_kernel(
    const float* __restrict__ a_in, const float* __restrict__ mean,
    const float* __restrict__ var, const float* __restrict__ gamma,
    const float* __restrict__ beta, float* __restrict__ a_out,
    float* __restrict__ ssum)
{
  __shared__ float sb[256];
  const int p = blockIdx.x;              // b*DP + c
  const int c = p % DP;
  const int tid = threadIdx.x;
  const float sc = rsqrtf(var[c] + BN_EPS) * gamma[c];
  const float mu = mean[c], be = beta[c];
  const float* ip = a_in + (size_t)p * HWp;
  float* op = a_out + (size_t)p * HWp;
  float loc = 0.f;
  for (int i = tid; i < HWp; i += 256) {
    const float t = (ip[i] - mu) * sc + be;
    const float sp = (t > 20.f) ? t : log1pf(__expf(t));
    op[i] = sp;
    loc += sp;
  }
  sb[tid] = loc; __syncthreads();
  for (int st = 128; st > 0; st >>= 1) {
    if (tid < st) sb[tid] += sb[tid + st];
    __syncthreads();
  }
  if (tid == 0) ssum[p] = sb[0];
}

// out = a / ssum[plane] * dis (may alias a)
__global__ void mul_norm_kernel(const float* __restrict__ a,
                                const float* __restrict__ ssum,
                                const float* __restrict__ dis,
                                float* __restrict__ out, long total)
{
  for (long i = (long)blockIdx.x * blockDim.x + threadIdx.x; i < total;
       i += (long)gridDim.x * blockDim.x) {
    const long p = i >> 12;
    out[i] = a[i] / ssum[p] * dis[i];
  }
}

// Final batchnorm + ReLU over [B, COUT, HW]
__global__ void bn_relu_kernel(const float* __restrict__ in,
                               const float* __restrict__ mean,
                               const float* __restrict__ var,
                               const float* __restrict__ gamma,
                               const float* __restrict__ beta,
                               float* __restrict__ out)
{
  const long total = (long)Bsz * COUT * HWp;
  for (long i = (long)blockIdx.x * blockDim.x + threadIdx.x; i < total;
       i += (long)gridDim.x * blockDim.x) {
    const int c = (int)((i >> 12) & (COUT - 1));
    const float t = (in[i] - mean[c]) * rsqrtf(var[c] + BN_EPS) * gamma[c] + beta[c];
    out[i] = fmaxf(t, 0.f);
  }
}

// ---------------------------------------------------------------------
extern "C" void kernel_launch(void* const* d_in, const int* in_sizes, int n_in,
                              void* d_out, int out_size, void* d_ws, size_t ws_size,
                              hipStream_t stream)
{
  (void)in_sizes; (void)n_in; (void)out_size; (void)ws_size;
  const float* x        = (const float*)d_in[0];   // [B,256,64,64]
  const float* pre_w    = (const float*)d_in[1];   // [512,256]
  const float* pre_b    = (const float*)d_in[2];
  const float* sigma    = (const float*)d_in[3];
  const float* offsets  = (const float*)d_in[4];   // [64,2]
  const float* atten_w  = (const float*)d_in[5];
  const float* atten_b  = (const float*)d_in[6];
  const float* atten_g  = (const float*)d_in[7];
  const float* atten_be = (const float*)d_in[8];
  const float* post_w   = (const float*)d_in[9];   // [256,512]
  const float* post_b   = (const float*)d_in[10];
  const float* bn_g     = (const float*)d_in[11];
  const float* bn_b     = (const float*)d_in[12];
  float* out = (float*)d_out;

  char* ws = (char*)d_ws;
  const size_t S_DP = (size_t)Bsz * DP * HWp * sizeof(float);    // 128 MB
  const size_t S_CO = (size_t)Bsz * COUT * HWp * sizeof(float);  // 64 MB
  float* buf0 = (float*)ws;                       // displaced features / pre-conv
  float* buf1 = (float*)(ws + S_DP);              // pooled / attention
  float* buf2 = (float*)(ws + 2 * S_DP);          // out_skip [B,256,HW]
  float* stat = (float*)(ws + 2 * S_DP + S_CO);
  float* mean_a = stat;                 // 512
  float* var_a  = stat + 512;           // 512
  float* ssum   = stat + 1024;          // 8192
  float* mean_o = stat + 1024 + 8192;   // 256
  float* var_o  = mean_o + 256;         // 256

  const dim3 blk(256);
  const int MBLKS = (Bsz * HWp) / BLK_M;          // 512

  // 1) out_pre = conv1x1(x, pre_w) + pre_b            -> buf0
  gemm_bf16_kernel<<<dim3(MBLKS, DP / BLK_N), blk, 0, stream>>>(
      x, pre_w, pre_b, nullptr, buf0, CIN, DP);
  // 2) dynamic gaussian pool (exp/log space)          -> buf1
  dyn_pool_kernel<<<Bsz * DP, blk, 0, stream>>>(buf0, sigma, buf1);
  // 3) displace (bilinear per-channel shift)          -> buf0
  displace_kernel<<<4096, blk, 0, stream>>>(buf1, offsets, buf0);
  // 4) attention conv1x1                              -> buf1
  gemm_bf16_kernel<<<dim3(MBLKS, DP / BLK_N), blk, 0, stream>>>(
      x, atten_w, atten_b, nullptr, buf1, CIN, DP);
  // 5) batchnorm stats over attention pre-activations
  chan_stats_kernel<<<DP, blk, 0, stream>>>(buf1, DP, mean_a, var_a);
  // 6) bn + softplus + spatial sums (in place)
  atten_norm_kernel<<<Bsz * DP, blk, 0, stream>>>(
      buf1, mean_a, var_a, atten_g, atten_be, buf1, ssum);
  // 7) space-normalize and multiply by displaced feat (in place)
  mul_norm_kernel<<<4096, blk, 0, stream>>>(
      buf1, ssum, buf0, buf1, (long)Bsz * DP * HWp);
  // 8) post conv1x1 + bias + residual skip            -> buf2
  gemm_bf16_kernel<<<dim3(MBLKS, COUT / BLK_N), blk, 0, stream>>>(
      buf1, post_w, post_b, x, buf2, DP, COUT);
  // 9) final batchnorm stats
  chan_stats_kernel<<<COUT, blk, 0, stream>>>(buf2, COUT, mean_o, var_o);
  // 10) bn + relu                                     -> d_out
  bn_relu_kernel<<<4096, blk, 0, stream>>>(buf2, mean_o, var_o, bn_g, bn_b, out);
}
